// SwinTransformer_17428977287304
// MI455X (gfx1250) — compile-verified, matrix operand order checked by reading, not run on hardware
//
#include <hip/hip_runtime.h>
#include <hip/hip_bf16.h>

// ---------------------------------------------------------------------------
// Swin block, MI455X (gfx1250, wave32). GEMMs on v_wmma_f32_16x16x32_f16 with
// A-stationary register blocking (activations read once) and LDS-staged
// weights. Weight staging uses the gfx1250 async global->LDS datapath
// (GLOBAL_LOAD_ASYNC_TO_LDS_B128 / ASYNCcnt) when the builtin is available.
// B=32 H=W=128 C=128 WS=2 SS=1 NH=8 HD=16 SZ=4 NW=4096 ; tokens = 524288.
// ---------------------------------------------------------------------------

typedef __attribute__((ext_vector_type(16))) _Float16 v16h;
typedef __attribute__((ext_vector_type(8)))  _Float16 v8h;
typedef __attribute__((ext_vector_type(4)))  _Float16 v4h;
typedef __attribute__((ext_vector_type(8)))  float    v8f;
typedef __attribute__((ext_vector_type(4)))  int      v4i;

#define TOKENS   524288      // B*H*W
#define NWIN     131072      // B*NW
#define MTILES   32768       // TOKENS/16

#if defined(__has_builtin)
#if __has_builtin(__builtin_amdgcn_global_load_async_to_lds_b128)
#define HAVE_ASYNC_LDS 1
#endif
#endif

static __device__ __forceinline__ int lane_id() { return threadIdx.x & 31; }
static __device__ __forceinline__ int wave_id() { return threadIdx.x >> 5; }

// Region id for the shifted-window attention mask (H=W=128, WS=2, SS=1)
static __device__ __forceinline__ int regf(int t) {
  return t < 126 ? 0 : (t == 126 ? 1 : 2);
}

// gather/scatter map: window-token index t -> flat spatial row in [B, H*W]
static __device__ __forceinline__ size_t token_row(int t) {
  int n  = t >> 2, s = t & 3;
  int bb = n >> 12, w = n & 4095;
  int wy = w >> 6,  wx = w & 63;
  int r  = 2 * wy + (s >> 1);
  int c  = 2 * wx + (s & 1);
  return (size_t)bb * 16384 + (size_t)((r + 1) & 127) * 128 + (size_t)((c + 1) & 127);
}

// Fragment loader for v_wmma_f32_16x16x32_f16. Row-major f16 [*, ld], works
// for both global and LDS pointers (inlined, addrspace inferred).
// 16-bit A-matrix layout (ISA 7.12.2): lane 0-15 -> M=lane, K 0..7 / 16..23
// in vgprs 0-3 / 4-7; lanes 16-31 same M, K +8. B mirrored on W^T rows.
static __device__ __forceinline__ v16h load_frag(const _Float16* p,
                                                 int ld, int row0, int k0) {
  int l = lane_id();
  const _Float16* base = p + (size_t)(row0 + (l & 15)) * ld + k0 + ((l >> 4) << 3);
  v8h lo = *(const v8h*)(base);
  v8h hi = *(const v8h*)(base + 16);
  v16h f;
#pragma unroll
  for (int i = 0; i < 8; ++i) { f[i] = lo[i]; f[i + 8] = hi[i]; }
  return f;
}

static __device__ __forceinline__ float wave_sum(float v) {
#pragma unroll
  for (int m = 16; m > 0; m >>= 1) v += __shfl_xor(v, m, 32);
  return v;
}

// Stage a [N,K] f16 weight matrix into LDS (whole workgroup cooperates).
// K*N is always a multiple of 2048 here, so no tail / no divergence.
template<int K, int N>
static __device__ __forceinline__ void stage_weights(const _Float16* __restrict__ Wt,
                                                     _Float16* s) {
  constexpr int CH = (K * N) / 8;          // 16-byte chunks
#if HAVE_ASYNC_LDS
  __attribute__((address_space(1))) v4i* gsrc =
      (__attribute__((address_space(1))) v4i*)Wt;
  __attribute__((address_space(3))) v4i* ldst =
      (__attribute__((address_space(3))) v4i*)s;
#pragma unroll 4
  for (int i = threadIdx.x; i < CH; i += 256)
    __builtin_amdgcn_global_load_async_to_lds_b128(gsrc + i, ldst + i, 0, 0);
#if __has_builtin(__builtin_amdgcn_s_wait_asynccnt)
  __builtin_amdgcn_s_wait_asynccnt(0);
#else
  asm volatile("s_wait_asynccnt 0x0" ::: "memory");
#endif
  __syncthreads();
#else
#pragma unroll 4
  for (int i = threadIdx.x; i < CH; i += 256)
    ((v8h*)s)[i] = ((const v8h*)Wt)[i];
  __syncthreads();
#endif
}

#define WMMA_F16(A_, B_, C_) \
  __builtin_amdgcn_wmma_f32_16x16x32_f16(false, (A_), false, (B_), (short)0, (C_), false, false)

// ---------------------------------------------------------------------------
// Weight cast + transpose: src f32 [K, N] -> dst f16 [N, K]
// ---------------------------------------------------------------------------
__global__ void k_cast_transpose(const float* __restrict__ src,
                                 _Float16* __restrict__ dst, int K, int N) {
  int idx = blockIdx.x * 256 + threadIdx.x;
  if (idx >= K * N) return;
  int k = idx / N, n = idx % N;
  dst[(size_t)n * K + k] = (_Float16)src[(size_t)k * N + n];
}

// ---------------------------------------------------------------------------
// LayerNorm over C=128, one wave per token; optional gather through the
// shift+window-partition map (LN1). Output f16 [TOKENS, 128].
// ---------------------------------------------------------------------------
__global__ void k_layernorm(const float* __restrict__ src,
                            const float* __restrict__ gamma,
                            const float* __restrict__ beta,
                            _Float16* __restrict__ dst, int gather) {
  int t = blockIdx.x * 8 + wave_id();
  int l = lane_id();
  size_t srow = gather ? token_row(t) : (size_t)t;
  float4 v = ((const float4*)(src + srow * 128))[l];
  float mean = wave_sum(v.x + v.y + v.z + v.w) * (1.0f / 128.0f);
  float dx = v.x - mean, dy = v.y - mean, dz = v.z - mean, dw = v.w - mean;
  float var = wave_sum(dx * dx + dy * dy + dz * dz + dw * dw) * (1.0f / 128.0f);
  float inv = rsqrtf(var + 1e-5f);
  float4 g = ((const float4*)gamma)[l];
  float4 b = ((const float4*)beta)[l];
  v4h o;
  o[0] = (_Float16)(dx * inv * g.x + b.x);
  o[1] = (_Float16)(dy * inv * g.y + b.y);
  o[2] = (_Float16)(dz * inv * g.z + b.z);
  o[3] = (_Float16)(dw * inv * g.w + b.w);
  *(v4h*)(dst + (size_t)t * 128 + l * 4) = o;
}

// ---------------------------------------------------------------------------
// QKV GEMM: A f16 [TOKENS,128] x Wt f16 [384,128] + bias -> f16 [TOKENS,384]
// A-stationary: one wave owns a 16-row block, K=128 held in 4 fragments,
// sweeps all 384 output columns from LDS-resident weights.
// ---------------------------------------------------------------------------
__global__ __launch_bounds__(256) void k_qkv_gemm(const _Float16* __restrict__ A,
                                                  const _Float16* __restrict__ Wt,
                                                  const float* __restrict__ bias,
                                                  _Float16* __restrict__ out) {
  extern __shared__ _Float16 smem[];
  stage_weights<128, 384>(Wt, smem);
  int mt = blockIdx.x * 8 + wave_id();
  int row0 = mt * 16;
  if (mt + 128 < MTILES)
    __builtin_prefetch(A + (size_t)(row0 + 2048) * 128, 0, 1);
  v16h afr[4];
#pragma unroll
  for (int kk = 0; kk < 4; ++kk) afr[kk] = load_frag(A, 128, row0, kk * 32);
  int l = lane_id();
  int mbase = row0 + ((l >> 4) << 3);
#pragma unroll
  for (int n0 = 0; n0 < 384; n0 += 32) {
    v8f c0 = {}, c1 = {};
#pragma unroll
    for (int kk = 0; kk < 4; ++kk) {
      v16h b0 = load_frag(smem, 128, n0, kk * 32);
      v16h b1 = load_frag(smem, 128, n0 + 16, kk * 32);
      c0 = WMMA_F16(afr[kk], b0, c0);
      c1 = WMMA_F16(afr[kk], b1, c1);
    }
    int col0 = n0 + (l & 15), col1 = col0 + 16;
    float bz0 = bias[col0], bz1 = bias[col1];
#pragma unroll
    for (int r = 0; r < 8; ++r) {
      size_t row = (size_t)(mbase + r) * 384;
      out[row + col0] = (_Float16)(c0[r] + bz0);
      out[row + col1] = (_Float16)(c1[r] + bz1);
    }
  }
}

// ---------------------------------------------------------------------------
// Windowed attention: 1 wave per window, lane = (head h = l>>2, query i = l&3).
// qkv f16 [TOKENS, 384], column layout (sel, head, d). Output f16 [TOKENS,128].
// ---------------------------------------------------------------------------
__global__ void k_attn(const _Float16* __restrict__ qkv,
                       const float* __restrict__ rpb,
                       _Float16* __restrict__ o) {
  int n = blockIdx.x * 8 + wave_id();       // window id
  int l = lane_id();
  int h = l >> 2, i = l & 3;
  int w  = n & 4095;
  int wy = w >> 6, wx = w & 63;
  int regi = regf(2 * wy + (i >> 1)) * 3 + regf(2 * wx + (i & 1));
  size_t base = (size_t)n * 4 * 384;

  const v8h* qp = (const v8h*)(qkv + base + (size_t)i * 384 + h * 16);
  v8h q0 = qp[0], q1 = qp[1];

  float a[4];
#pragma unroll
  for (int j = 0; j < 4; ++j) {
    const v8h* kp = (const v8h*)(qkv + base + (size_t)j * 384 + 128 + h * 16);
    v8h kk0 = kp[0], kk1 = kp[1];
    float d = 0.f;
#pragma unroll
    for (int e = 0; e < 8; ++e)
      d += (float)q0[e] * (float)kk0[e] + (float)q1[e] * (float)kk1[e];
    int idx  = 3 * ((i >> 1) - (j >> 1) + 1) + ((i & 1) - (j & 1) + 1);
    int regj = regf(2 * wy + (j >> 1)) * 3 + regf(2 * wx + (j & 1));
    a[j] = 0.25f * d + rpb[idx * 8 + h] + (regi != regj ? -100.0f : 0.0f);
  }
  float mx = fmaxf(fmaxf(a[0], a[1]), fmaxf(a[2], a[3]));
  float p[4], s = 0.f;
#pragma unroll
  for (int j = 0; j < 4; ++j) { p[j] = __expf(a[j] - mx); s += p[j]; }
  float inv = 1.0f / s;

  float acc[16];
#pragma unroll
  for (int e = 0; e < 16; ++e) acc[e] = 0.f;
#pragma unroll
  for (int j = 0; j < 4; ++j) {
    const v8h* vp = (const v8h*)(qkv + base + (size_t)j * 384 + 256 + h * 16);
    v8h v0 = vp[0], v1 = vp[1];
    float pj = p[j] * inv;
#pragma unroll
    for (int e = 0; e < 8; ++e) {
      acc[e]     += pj * (float)v0[e];
      acc[e + 8] += pj * (float)v1[e];
    }
  }
  v8h o0, o1;
#pragma unroll
  for (int e = 0; e < 8; ++e) { o0[e] = (_Float16)acc[e]; o1[e] = (_Float16)acc[e + 8]; }
  v8h* op = (v8h*)(o + ((size_t)n * 4 + i) * 128 + h * 16);
  op[0] = o0; op[1] = o1;
}

// ---------------------------------------------------------------------------
// Proj GEMM + window-reverse + unshift + residual(x): -> h fp32 [TOKENS,128]
// ---------------------------------------------------------------------------
__global__ __launch_bounds__(256) void k_proj_gemm(const _Float16* __restrict__ A,
                                                   const _Float16* __restrict__ Wt,
                                                   const float* __restrict__ bias,
                                                   const float* __restrict__ x,
                                                   float* __restrict__ hout) {
  extern __shared__ _Float16 smem[];
  stage_weights<128, 128>(Wt, smem);
  int mt = blockIdx.x * 8 + wave_id();
  int row0 = mt * 16;
  if (mt + 128 < MTILES)
    __builtin_prefetch(A + (size_t)(row0 + 2048) * 128, 0, 1);
  v16h afr[4];
#pragma unroll
  for (int kk = 0; kk < 4; ++kk) afr[kk] = load_frag(A, 128, row0, kk * 32);
  int l = lane_id();
  int mbase = row0 + ((l >> 4) << 3);
#pragma unroll
  for (int n0 = 0; n0 < 128; n0 += 32) {
    v8f c0 = {}, c1 = {};
#pragma unroll
    for (int kk = 0; kk < 4; ++kk) {
      v16h b0 = load_frag(smem, 128, n0, kk * 32);
      v16h b1 = load_frag(smem, 128, n0 + 16, kk * 32);
      c0 = WMMA_F16(afr[kk], b0, c0);
      c1 = WMMA_F16(afr[kk], b1, c1);
    }
    int col0 = n0 + (l & 15), col1 = col0 + 16;
    float bz0 = bias[col0], bz1 = bias[col1];
#pragma unroll
    for (int r = 0; r < 8; ++r) {
      size_t grow = token_row(mbase + r) * 128;
      hout[grow + col0] = x[grow + col0] + c0[r] + bz0;
      hout[grow + col1] = x[grow + col1] + c1[r] + bz1;
    }
  }
}

// ---------------------------------------------------------------------------
// MLP1 GEMM + exact GELU: A f16 [TOKENS,128] x w1t [256,128] -> f16 [TOKENS,256]
// ---------------------------------------------------------------------------
__global__ __launch_bounds__(256) void k_mlp1_gemm(const _Float16* __restrict__ A,
                                                   const _Float16* __restrict__ Wt,
                                                   const float* __restrict__ bias,
                                                   _Float16* __restrict__ g) {
  extern __shared__ _Float16 smem[];
  stage_weights<128, 256>(Wt, smem);
  int mt = blockIdx.x * 8 + wave_id();
  int row0 = mt * 16;
  if (mt + 128 < MTILES)
    __builtin_prefetch(A + (size_t)(row0 + 2048) * 128, 0, 1);
  v16h afr[4];
#pragma unroll
  for (int kk = 0; kk < 4; ++kk) afr[kk] = load_frag(A, 128, row0, kk * 32);
  int l = lane_id();
  int mbase = row0 + ((l >> 4) << 3);
#pragma unroll
  for (int n0 = 0; n0 < 256; n0 += 32) {
    v8f c0 = {}, c1 = {};
#pragma unroll
    for (int kk = 0; kk < 4; ++kk) {
      v16h b0 = load_frag(smem, 128, n0, kk * 32);
      v16h b1 = load_frag(smem, 128, n0 + 16, kk * 32);
      c0 = WMMA_F16(afr[kk], b0, c0);
      c1 = WMMA_F16(afr[kk], b1, c1);
    }
    int col0 = n0 + (l & 15), col1 = col0 + 16;
    float bz0 = bias[col0], bz1 = bias[col1];
#pragma unroll
    for (int r = 0; r < 8; ++r) {
      size_t row = (size_t)(mbase + r) * 256;
      float v0 = c0[r] + bz0;
      float v1 = c1[r] + bz1;
      v0 = 0.5f * v0 * (1.0f + erff(v0 * 0.70710678118f));
      v1 = 0.5f * v1 * (1.0f + erff(v1 * 0.70710678118f));
      g[row + col0] = (_Float16)v0;
      g[row + col1] = (_Float16)v1;
    }
  }
}

// ---------------------------------------------------------------------------
// MLP2 GEMM (K=256) + residual(h): -> d_out fp32 [TOKENS,128]
// ---------------------------------------------------------------------------
__global__ __launch_bounds__(256) void k_mlp2_gemm(const _Float16* __restrict__ A,
                                                   const _Float16* __restrict__ Wt,
                                                   const float* __restrict__ bias,
                                                   const float* __restrict__ hres,
                                                   float* __restrict__ out) {
  extern __shared__ _Float16 smem[];
  stage_weights<256, 128>(Wt, smem);
  int mt = blockIdx.x * 8 + wave_id();
  int row0 = mt * 16;
  if (mt + 128 < MTILES)
    __builtin_prefetch(A + (size_t)(row0 + 2048) * 256, 0, 1);
  v16h afr[8];
#pragma unroll
  for (int kk = 0; kk < 8; ++kk) afr[kk] = load_frag(A, 256, row0, kk * 32);
  int l = lane_id();
  int mbase = row0 + ((l >> 4) << 3);
#pragma unroll
  for (int n0 = 0; n0 < 128; n0 += 32) {
    v8f c0 = {}, c1 = {};
#pragma unroll
    for (int kk = 0; kk < 8; ++kk) {
      v16h b0 = load_frag(smem, 256, n0, kk * 32);
      v16h b1 = load_frag(smem, 256, n0 + 16, kk * 32);
      c0 = WMMA_F16(afr[kk], b0, c0);
      c1 = WMMA_F16(afr[kk], b1, c1);
    }
    int col0 = n0 + (l & 15), col1 = col0 + 16;
    float bz0 = bias[col0], bz1 = bias[col1];
#pragma unroll
    for (int r = 0; r < 8; ++r) {
      size_t row = (size_t)(mbase + r) * 128;
      out[row + col0] = hres[row + col0] + c0[r] + bz0;
      out[row + col1] = hres[row + col1] + c1[r] + bz1;
    }
  }
}

// ---------------------------------------------------------------------------
// Host launcher. Workspace layout (bytes):
//   [0        , 262144)   : f16 transposed weights (qkv_wt, proj_wt, w1t, w2t)
//   [262144   , +128 MB)  : hw f16  [TOKENS,128]  (reused as attention output o)
//   [+        , +384 MB)  : qkv f16 [TOKENS,384]  (reused: LN2 out m + MLP hidden g)
//   [+        , +256 MB)  : h  fp32 [TOKENS,128]  (first residual)
// ---------------------------------------------------------------------------
extern "C" void kernel_launch(void* const* d_in, const int* in_sizes, int n_in,
                              void* d_out, int out_size, void* d_ws, size_t ws_size,
                              hipStream_t stream) {
  const float* x      = (const float*)d_in[0];
  const float* qkv_w  = (const float*)d_in[1];
  const float* qkv_b  = (const float*)d_in[2];
  const float* proj_w = (const float*)d_in[3];
  const float* proj_b = (const float*)d_in[4];
  const float* rpb    = (const float*)d_in[5];
  const float* gamma1 = (const float*)d_in[6];
  const float* beta1  = (const float*)d_in[7];
  const float* gamma2 = (const float*)d_in[8];
  const float* beta2  = (const float*)d_in[9];
  const float* w1     = (const float*)d_in[10];
  const float* b1     = (const float*)d_in[11];
  const float* w2     = (const float*)d_in[12];
  const float* b2     = (const float*)d_in[13];

  char* ws = (char*)d_ws;
  _Float16* qkv_wt  = (_Float16*)(ws + 0);
  _Float16* proj_wt = (_Float16*)(ws + 98304);
  _Float16* w1t     = (_Float16*)(ws + 131072);
  _Float16* w2t     = (_Float16*)(ws + 196608);
  _Float16* hw      = (_Float16*)(ws + 262144);                  // 128 MB (also o)
  _Float16* qkvb    = (_Float16*)(ws + 262144 + 134217728ULL);   // 384 MB
  _Float16* mbuf    = qkvb;                                      // LN2 out (128 MB)
  _Float16* gbuf    = (_Float16*)((char*)qkvb + 134217728ULL);   // MLP hidden (256 MB)
  float*    hres    = (float*)(ws + 262144 + 134217728ULL + 402653184ULL); // 256 MB
  float*    out     = (float*)d_out;

  k_cast_transpose<<<(128 * 384 + 255) / 256, 256, 0, stream>>>(qkv_w, qkv_wt, 128, 384);
  k_cast_transpose<<<(128 * 128 + 255) / 256, 256, 0, stream>>>(proj_w, proj_wt, 128, 128);
  k_cast_transpose<<<(128 * 256 + 255) / 256, 256, 0, stream>>>(w1, w1t, 128, 256);
  k_cast_transpose<<<(256 * 128 + 255) / 256, 256, 0, stream>>>(w2, w2t, 256, 128);

  // LN1 + shift + window partition (gather) -> hw f16
  k_layernorm<<<TOKENS / 8, 256, 0, stream>>>(x, gamma1, beta1, hw, 1);
  // QKV: 4096 blocks x 8 waves, each wave owns one 16-row block, all 384 cols
  k_qkv_gemm<<<MTILES / 8, 256, 98304, stream>>>(hw, qkv_wt, qkv_b, qkvb);
  // attention, write o into hw region
  k_attn<<<NWIN / 8, 256, 0, stream>>>(qkvb, rpb, hw);
  // proj + reverse/unshift + residual -> hres
  k_proj_gemm<<<MTILES / 8, 256, 32768, stream>>>(hw, proj_wt, proj_b, x, hres);
  // LN2 -> mbuf f16
  k_layernorm<<<TOKENS / 8, 256, 0, stream>>>(hres, gamma2, beta2, mbuf, 0);
  // MLP1 (+GELU) -> gbuf
  k_mlp1_gemm<<<MTILES / 8, 256, 65536, stream>>>(mbuf, w1t, b1, gbuf);
  // MLP2 + residual -> out
  k_mlp2_gemm<<<MTILES / 8, 256, 65536, stream>>>(gbuf, w2t, b2, hres, out);

  (void)in_sizes; (void)n_in; (void)out_size; (void)ws_size;
}
